// CrossAttention_5961414607179
// MI455X (gfx1250) — matrix operand order repaired and checked by reading, not compiled, not run on hardware
//
#include <hip/hip_runtime.h>
#include <hip/hip_bf16.h>

typedef __bf16 bf16_t;
typedef __attribute__((ext_vector_type(16))) __bf16 v16bf;
typedef __attribute__((ext_vector_type(8)))  __bf16 v8bf;
typedef __attribute__((ext_vector_type(8)))  float  v8f;
typedef int v4i __attribute__((vector_size(16)));   // matches builtin proto

#define BATCH 4
#define CCH   256
#define DKQ   32
#define NSEQ  4096
#define BN    64        // key-chunk size

#if defined(__has_builtin)
#if __has_builtin(__builtin_amdgcn_global_load_async_to_lds_b128)
#define HAVE_ASYNC 1
#endif
#endif

// 16-byte global -> LDS copy (async CDNA5 path when available)
__device__ __forceinline__ void cp16_g2l(void* ldst, const void* gsrc) {
#ifdef HAVE_ASYNC
    __builtin_amdgcn_global_load_async_to_lds_b128(
        (__attribute__((address_space(1))) v4i*)(v4i*)const_cast<void*>(gsrc),
        (__attribute__((address_space(3))) v4i*)(v4i*)ldst,
        0, 0);
#else
    *(v8bf*)ldst = *(const v8bf*)gsrc;
#endif
}

// DPP16 lane swap (pure VALU, no LDS round-trip)
template <int CTRL>
__device__ __forceinline__ float dpp_swap(float v) {
    int x = __builtin_bit_cast(int, v);
    int y = __builtin_amdgcn_update_dpp(x, x, CTRL, 0xF, 0xF, true);
    return __builtin_bit_cast(float, y);
}
// full reduction over each 16-lane row: xor1, xor2, xor7, xor15 (GF(2)^4 basis)
__device__ __forceinline__ float rowmax16(float v) {
    v = fmaxf(v, dpp_swap<0xB1>(v));   // quad_perm(1,0,3,2) : xor 1
    v = fmaxf(v, dpp_swap<0x4E>(v));   // quad_perm(2,3,0,1) : xor 2
    v = fmaxf(v, dpp_swap<0x141>(v));  // row_half_mirror    : xor 7
    v = fmaxf(v, dpp_swap<0x140>(v));  // row_mirror         : xor 15
    return v;
}
__device__ __forceinline__ float rowsum16(float v) {
    v += dpp_swap<0xB1>(v);
    v += dpp_swap<0x4E>(v);
    v += dpp_swap<0x141>(v);
    v += dpp_swap<0x140>(v);
    return v;
}

// ---------------------------------------------------------------------------
// Pass 1: q/k/v projections (1x1 convs).
// ---------------------------------------------------------------------------
__global__ __launch_bounds__(256) void proj_qkv(
    const float* __restrict__ x,
    const float* __restrict__ Wq, const float* __restrict__ bq,
    const float* __restrict__ Wk, const float* __restrict__ bk,
    const float* __restrict__ Wv, const float* __restrict__ bv,
    bf16_t* __restrict__ qws, bf16_t* __restrict__ kws, bf16_t* __restrict__ vws)
{
    __shared__ float xs[CCH * 32];
    const int b  = blockIdx.x >> 7;
    const int n0 = (blockIdx.x & 127) * 32;
    const int t  = threadIdx.x;
    const float* xb = x + (size_t)b * CCH * NSEQ;

    for (int idx = t; idx < CCH * 32; idx += 256) {
        int c = idx >> 5, nn = idx & 31;
        xs[idx] = xb[(size_t)c * NSEQ + n0 + nn];
    }
    __syncthreads();

    const float scale = 0.17677669529663689f;   // 1/sqrt(32)

    for (int idx = t; idx < 32 * 32; idx += 256) {
        int nn = idx >> 5, d = idx & 31;
        const float* wq = Wq + d * CCH;
        const float* wk = Wk + d * CCH;
        float sq = 0.f, sk = 0.f;
        for (int c = 0; c < CCH; ++c) {
            float xv = xs[c * 32 + nn];
            sq += wq[c] * xv;
            sk += wk[c] * xv;
        }
        sq = (sq + bq[d]) * scale;
        sk = sk + bk[d];
        size_t o = ((size_t)b * NSEQ + n0 + nn) * DKQ + d;
        qws[o] = (bf16_t)sq;
        kws[o] = (bf16_t)sk;
    }
    for (int idx = t; idx < 32 * CCH; idx += 256) {
        int co = idx >> 5, nn = idx & 31;
        const float* wv = Wv + co * CCH;
        float s = 0.f;
        for (int c = 0; c < CCH; ++c) s += wv[c] * xs[c * 32 + nn];
        s += bv[co];
        vws[((size_t)b * CCH + co) * NSEQ + n0 + nn] = (bf16_t)s;
    }
}

// ---------------------------------------------------------------------------
// Pass 2: flash attention. 8 waves/block: wave w owns row-block (w>>1) and
// channel-half (w&1) -> per-wave acc is 16x128 f32 = 64 VGPRs (stays under
// the 256-VGPR direct window; no s_set_vgpr_msb churn). k/v chunks staged in
// LDS shared by all 8 waves, double-buffered via async global->LDS.
// ---------------------------------------------------------------------------
__global__ __launch_bounds__(256) void flash_attn(
    const bf16_t* __restrict__ qws, const bf16_t* __restrict__ kws,
    const bf16_t* __restrict__ vws, const float* __restrict__ x,
    float* __restrict__ out)
{
    __shared__ bf16_t Vl[2][CCH * BN];      // 64 KB: v chunk, [c][key]
    __shared__ bf16_t Kl[2][BN * DKQ];      //  8 KB: k chunk, [key][d]
    __shared__ bf16_t Plds[8][16 * BN];     // 16 KB: per-wave P staging

    const int t    = threadIdx.x;
    const int w    = t >> 5;
    const int lane = t & 31;
    const int rblk = blockIdx.x * 4 + (w >> 1);
    const int ch   = w & 1;                 // channel half: 0 -> c<128, 1 -> c>=128
    const int b    = rblk >> 8;             // all 8 waves share one batch
    const int m0   = (rblk & 255) * 16;

    const int half = lane >> 4;
    const int l16  = lane & 15;

    const bf16_t* kb = kws + (size_t)b * NSEQ * DKQ;
    const bf16_t* vb = vws + (size_t)b * CCH * NSEQ;

    // stage one 64-key chunk of k and v into LDS buffer `buf` (256 threads)
    auto stage = [&](int buf, int nc) {
        #pragma unroll
        for (int i0 = 0; i0 < 2048; i0 += 256) {
            int i = i0 + t;
            int c = i >> 3, s = i & 7;
            cp16_g2l(&Vl[buf][c * BN + s * 8], vb + (size_t)c * NSEQ + nc + s * 8);
        }
        cp16_g2l(&Kl[buf][t * 8], kb + (size_t)nc * DKQ + t * 8);
    };

    // A operand: q tile 16x32 (K = DKQ exactly); contiguous k = 16*half + s.
    const bf16_t* qp = qws + ((size_t)b * NSEQ + m0 + l16) * DKQ;
    v8bf qlo = *(const v8bf*)(qp + half * 16);
    v8bf qhi = *(const v8bf*)(qp + half * 16 + 8);
    v16bf aq;
    #pragma unroll
    for (int i = 0; i < 8; ++i) { aq[i] = qlo[i]; aq[8 + i] = qhi[i]; }

    const v8f zero = {0.f, 0.f, 0.f, 0.f, 0.f, 0.f, 0.f, 0.f};
    v8f acc[8];                              // 16 rows x 128 channels
    #pragma unroll
    for (int i = 0; i < 8; ++i) acc[i] = zero;
    float mrow[8], lrow[8];                  // lrow is LANE-LOCAL partial sum
    #pragma unroll
    for (int r = 0; r < 8; ++r) { mrow[r] = -3.0e38f; lrow[r] = 0.f; }

    bf16_t* pw = &Plds[w][0];

    // prologue: stage chunk 0
    stage(0, 0);
#ifdef HAVE_ASYNC
    asm volatile("s_wait_asynccnt 0" ::: "memory");
#endif
    __syncthreads();

    const int nchunks = NSEQ / BN;           // 64
    for (int ci = 0; ci < nchunks; ++ci) {
        const int cur = ci & 1;
        if (ci + 1 < nchunks) stage(1 - cur, (ci + 1) * BN);   // overlap copy

        // ---- energy: preload all 4 k tiles, then 4 S WMMAs ----
        v16bf bkt[4];
        #pragma unroll
        for (int nt = 0; nt < 4; ++nt)
            bkt[nt] = *(const v16bf*)(&Kl[cur][(nt * 16 + l16) * DKQ + half * 16]);
        v8f s[4];
        #pragma unroll
        for (int nt = 0; nt < 4; ++nt)
            s[nt] = __builtin_amdgcn_wmma_f32_16x16x32_bf16(false, aq, false, bkt[nt],
                                                            (short)0, zero, false, false);

        // ---- online softmax: DPP row-max, lane-local row-sum ----
        #pragma unroll
        for (int r = 0; r < 8; ++r) {
            float cm = fmaxf(fmaxf(s[0][r], s[1][r]), fmaxf(s[2][r], s[3][r]));
            cm = rowmax16(cm);
            float mn    = fmaxf(mrow[r], cm);
            float alpha = __expf(mrow[r] - mn);
            mrow[r] = mn;
            float e0 = __expf(s[0][r] - mn);
            float e1 = __expf(s[1][r] - mn);
            float e2 = __expf(s[2][r] - mn);
            float e3 = __expf(s[3][r] - mn);
            lrow[r] = lrow[r] * alpha + (e0 + e1 + e2 + e3);
            int m = half * 8 + r;            // C-layout row -> P row
            pw[m * BN + l16]      = (bf16_t)e0;
            pw[m * BN + 16 + l16] = (bf16_t)e1;
            pw[m * BN + 32 + l16] = (bf16_t)e2;
            pw[m * BN + 48 + l16] = (bf16_t)e3;
            #pragma unroll
            for (int ct = 0; ct < 8; ++ct) acc[ct][r] *= alpha;
        }
        asm volatile("s_wait_dscnt 0" ::: "memory");   // intra-wave P RAW

        // ---- O += P(16x64) @ V(64x16ch) for this wave's 8 channel tiles ----
        const bf16_t* vl0 = &Vl[cur][(size_t)(ch * 128 + l16) * BN + half * 16];
        #pragma unroll
        for (int sub = 0; sub < 2; ++sub) {
            const bf16_t* pr = pw + l16 * BN + sub * 32;   // A row m = l16
            v8bf plo = *(const v8bf*)(pr + half * 16);
            v8bf phi = *(const v8bf*)(pr + half * 16 + 8);
            v16bf pa;
            #pragma unroll
            for (int i = 0; i < 8; ++i) { pa[i] = plo[i]; pa[8 + i] = phi[i]; }

            // 2-deep software pipeline: ds_load for ct+1 ahead of WMMA for ct
            v16bf bcur = *(const v16bf*)(vl0 + sub * 32);
            #pragma unroll
            for (int ct = 0; ct < 8; ++ct) {
                v16bf bnext = bcur;
                if (ct < 7)
                    bnext = *(const v16bf*)(vl0 + (size_t)(ct + 1) * 16 * BN + sub * 32);
                acc[ct] = __builtin_amdgcn_wmma_f32_16x16x32_bf16(
                    false, pa, false, bcur, (short)0, acc[ct], false, false);
                bcur = bnext;
            }
        }

#ifdef HAVE_ASYNC
        asm volatile("s_wait_asynccnt 0" ::: "memory");    // next chunk landed
#endif
        __syncthreads();                                   // buffer handoff
    }

    // ---- epilogue: reduce deferred row sums, normalize, residual, store ----
    const float* xb = x   + (size_t)b * CCH * NSEQ;
    float*       ob = out + (size_t)b * CCH * NSEQ;
    #pragma unroll
    for (int r = 0; r < 8; ++r) {
        float lsum = rowsum16(lrow[r]);
        float inv  = 1.f / lsum;
        int m = m0 + half * 8 + r;
        #pragma unroll
        for (int ct = 0; ct < 8; ++ct) {
            int c = ch * 128 + ct * 16 + l16;
            ob[(size_t)c * NSEQ + m] = acc[ct][r] * inv + xb[(size_t)c * NSEQ + m];
        }
    }
}

extern "C" void kernel_launch(void* const* d_in, const int* in_sizes, int n_in,
                              void* d_out, int out_size, void* d_ws, size_t ws_size,
                              hipStream_t stream) {
    (void)in_sizes; (void)n_in; (void)out_size; (void)ws_size;
    const float* x  = (const float*)d_in[0];
    const float* Wq = (const float*)d_in[1];
    const float* bq = (const float*)d_in[2];
    const float* Wk = (const float*)d_in[3];
    const float* bk = (const float*)d_in[4];
    const float* Wv = (const float*)d_in[5];
    const float* bv = (const float*)d_in[6];

    bf16_t* qws = (bf16_t*)d_ws;                               // 1 MB
    bf16_t* kws = qws + (size_t)BATCH * NSEQ * DKQ;            // 1 MB
    bf16_t* vws = kws + (size_t)BATCH * NSEQ * DKQ;            // 8 MB

    proj_qkv<<<BATCH * (NSEQ / 32), 256, 0, stream>>>(
        x, Wq, bq, Wk, bk, Wv, bv, qws, kws, vws);

    flash_attn<<<(BATCH * NSEQ / 16) / 4, 256, 0, stream>>>(
        qws, kws, vws, x, (float*)d_out);
}
